// Ranker_88450556494672
// MI455X (gfx1250) — matrix operand order repaired and checked by reading, not compiled, not run on hardware
//
#include <hip/hip_runtime.h>
#include <math.h>

typedef __attribute__((ext_vector_type(16))) _Float16 v16h;
typedef __attribute__((ext_vector_type(8)))  float    v8f;

// ---------------------------------------------------------------- helpers ---

__device__ __forceinline__ float sigmf(float x) { return 1.f / (1.f + __expf(-x)); }

__device__ __forceinline__ v8f splat8(float x) {
  v8f r;
#pragma unroll
  for (int i = 0; i < 8; ++i) r[i] = x;
  return r;
}

// A-matrix 16x32 f16 tile per CDNA5 WMMA striping (ISA 7.12.2):
// lane%16 = M row, lane/16 = half. Lane holds K = lh*8 + [0..7] and
// 16 + lh*8 + [0..7]  (two 16-byte runs at +0 and +32 bytes).
__device__ __forceinline__ v16h load_tileA16x32(const _Float16* base, size_t rowStride, int k0) {
  const int lane = threadIdx.x & 31;
  const int r    = lane & 15;
  const int lh   = lane >> 4;
  const _Float16* p = base + (size_t)r * rowStride + (size_t)(k0 + lh * 8);
  v16h v;
#pragma unroll
  for (int e = 0; e < 8; ++e) v[e] = p[e];
#pragma unroll
  for (int e = 0; e < 8; ++e) v[8 + e] = p[16 + e];
  return v;
}

// B-matrix 32x16 f16 tile per CDNA5 WMMA striping (ISA 7.12.4 B layout):
// lane%16 = N column, lane/16 = half. Lane holds K = lh*8 + [0..7] and
// 16 + lh*8 + [0..7] (same as A). B[k][n] = W[(n0+n)*ld + k]
// (W row-major with rows = output columns), so pass base = W + n0*ld.
__device__ __forceinline__ v16h load_tileB16x32(const _Float16* base, size_t rowStride, int k0) {
  const int lane = threadIdx.x & 31;
  const int n    = lane & 15;
  const int lh   = lane >> 4;
  const _Float16* p = base + (size_t)n * rowStride + (size_t)(k0 + lh * 8);
  v16h v;
#pragma unroll
  for (int e = 0; e < 8; ++e) v[e] = p[e];
#pragma unroll
  for (int e = 0; e < 8; ++e) v[8 + e] = p[16 + e];
  return v;
}

__device__ __forceinline__ v8f wmma_f16(v16h a, v16h b, v8f c) {
  // (neg_a, A, neg_b, B, c_mod, C, reuse_a, reuse_b)
  return __builtin_amdgcn_wmma_f32_16x16x32_f16(false, a, false, b, (short)0, c, false, false);
}

// ---------------------------------------------------- small prep kernels ----

// f32 -> f16 with column zero-padding (rows x cin) -> (rows x cout)
__global__ void cvt_pad_kernel(const float* __restrict__ in, _Float16* __restrict__ out,
                               int rows, int cin, int cout) {
  size_t i = (size_t)blockIdx.x * 256 + threadIdx.x;
  if (i >= (size_t)rows * cout) return;
  int r = (int)(i / cout), c = (int)(i % cout);
  out[i] = (c < cin) ? (_Float16)in[(size_t)r * cin + c] : (_Float16)0.f;
}

__global__ void bias_add_kernel(const float* __restrict__ a, const float* __restrict__ b,
                                float* __restrict__ o, int n) {
  int i = blockIdx.x * 256 + threadIdx.x;
  if (i < n) o[i] = a[i] + b[i];
}

__global__ void zero_f16_kernel(_Float16* __restrict__ p, size_t n) {
  size_t i = (size_t)blockIdx.x * 256 + threadIdx.x;
  if (i < n) p[i] = (_Float16)0.f;
}

// Gather + L2-normalize embedding rows on the fly (row 0 -> 0). Out: (ntok, 320) f16.
__global__ void gather_norm_kernel(const int* __restrict__ idx, const float* __restrict__ emb,
                                   _Float16* __restrict__ out) {
  const int tok = blockIdx.x;
  const int tid = threadIdx.x;  // 128
  __shared__ float red[128];
  const int row = idx[tok];
  float ss = 0.f;
  for (int c = tid; c < 300; c += 128) {
    float v = emb[(size_t)row * 300 + c];
    ss += v * v;
  }
  red[tid] = ss;
  __syncthreads();
  for (int st = 64; st > 0; st >>= 1) {
    if (tid < st) red[tid] += red[tid + st];
    __syncthreads();
  }
  const float inv = (row == 0) ? 0.f : 1.f / fmaxf(sqrtf(red[0]), 1e-12f);
  for (int c = tid; c < 320; c += 128) {
    float v = (c < 300) ? emb[(size_t)row * 300 + c] * inv : 0.f;
    out[(size_t)tok * 320 + c] = (_Float16)v;
  }
}

// ------------------------------------------------------- WMMA LSTM kernel ---
// Each block: 16 sequences, 8 waves. Wave w owns hidden cols [w*H/8, (w+1)*H/8)
// and computes i/f/g/o gate tiles for its own columns -> no gate exchange.
// Cell state in registers; hidden state double-buffered f16 in LDS.
template <int DIN, int H, int CTPW, bool MAXPOOL>
__global__ __launch_bounds__(256) void lstm_wmma_kernel(
    const _Float16* __restrict__ X,    // (Nseq, T, DIN) f16, DIN mult of 32
    const _Float16* __restrict__ Wih,  // (4H, DIN) f16
    const _Float16* __restrict__ Whh,  // (4H, H) f16
    const float* __restrict__ bias,    // (4H) = bih + bhh
    int T, int reverse,
    float* __restrict__ outF,          // MAXPOOL: (Nseq, ldOut)
    _Float16* __restrict__ outH,       // !MAXPOOL: (Nseq, tPad, ldOut) @ coloff
    int ldOut, int coloff, int tPad) {
  __shared__ _Float16 hbuf[2][16][H];
  const int tid = threadIdx.x;
  const int wave = tid >> 5;
  const int lane = tid & 31;
  const int nl = lane & 15;
  const int lh = lane >> 4;
  const int seq0 = blockIdx.x * 16;
  const int colsPerWave = H / 8;

  for (int i = tid; i < 16 * H; i += 256) (&hbuf[0][0][0])[i] = (_Float16)0.f;
  __syncthreads();

  v8f cstate[CTPW];
  v8f hmax[CTPW];
#pragma unroll
  for (int ct = 0; ct < CTPW; ++ct) { cstate[ct] = splat8(0.f); hmax[ct] = splat8(-1e30f); }

  for (int ti = 0; ti < T; ++ti) {
    const int ts = reverse ? (T - 1 - ti) : ti;
    const int rd = ti & 1, wr = rd ^ 1;
    const _Float16* Xb = X + (size_t)seq0 * T * DIN + (size_t)ts * DIN;
#pragma unroll
    for (int ct = 0; ct < CTPW; ++ct) {
      const int colbase = wave * colsPerWave + ct * 16;
      v8f acc[4];
#pragma unroll
      for (int g = 0; g < 4; ++g) acc[g] = splat8(bias[g * H + colbase + nl]);
      // G += X_t @ Wih^T
      for (int k0 = 0; k0 < DIN; k0 += 32) {
        v16h a = load_tileA16x32(Xb, (size_t)T * DIN, k0);
#pragma unroll
        for (int g = 0; g < 4; ++g) {
          v16h b = load_tileB16x32(Wih + (size_t)(g * H + colbase) * DIN, DIN, k0);
          acc[g] = wmma_f16(a, b, acc[g]);
        }
      }
      // G += h_{t-1} @ Whh^T
      for (int k0 = 0; k0 < H; k0 += 32) {
        v16h a = load_tileA16x32(&hbuf[rd][0][0], H, k0);
#pragma unroll
        for (int g = 0; g < 4; ++g) {
          v16h b = load_tileB16x32(Whh + (size_t)(g * H + colbase) * H, H, k0);
          acc[g] = wmma_f16(a, b, acc[g]);
        }
      }
      // i,f,g,o -> c,h  (element j at lane: m = j + 8*lh, n = colbase + nl)
#pragma unroll
      for (int j = 0; j < 8; ++j) {
        float ig = sigmf(acc[0][j]);
        float fg = sigmf(acc[1][j]);
        float gg = tanhf(acc[2][j]);
        float og = sigmf(acc[3][j]);
        float c = fg * cstate[ct][j] + ig * gg;
        cstate[ct][j] = c;
        float h = og * tanhf(c);
        int m = j + 8 * lh;
        hbuf[wr][m][colbase + nl] = (_Float16)h;
        if constexpr (MAXPOOL) {
          hmax[ct][j] = fmaxf(hmax[ct][j], h);
        } else {
          if (outH)
            outH[((size_t)(seq0 + m) * tPad + ts) * ldOut + coloff + colbase + nl] = (_Float16)h;
          if (outF)
            outF[((size_t)(seq0 + m) * tPad + ts) * ldOut + coloff + colbase + nl] = h;
        }
      }
    }
    __syncthreads();
  }
  if constexpr (MAXPOOL) {
#pragma unroll
    for (int ct = 0; ct < CTPW; ++ct) {
      const int colbase = wave * colsPerWave + ct * 16;
#pragma unroll
      for (int j = 0; j < 8; ++j) {
        int m = j + 8 * lh;
        outF[(size_t)(seq0 + m) * ldOut + coloff + colbase + nl] = hmax[ct][j];
      }
    }
  }
}

// --------------------------------------------------------- attention GEMMs --

// alig[b,k,q] = sum_d s_enc[b,k,d] * q_enc[b,q,d]   (q padded to 112 with zero rows)
__global__ __launch_bounds__(256) void attn_score_kernel(const _Float16* __restrict__ s_enc_h,
                                                         const _Float16* __restrict__ q_enc_h,
                                                         float* __restrict__ alig) {
  const int b = blockIdx.y, mt = blockIdx.x;
  const int wave = threadIdx.x >> 5;
  if (wave >= 7) return;  // 7 N-tiles (112 q cols); whole-wave exit keeps EXEC full
  const int lane = threadIdx.x & 31, nl = lane & 15, lh = lane >> 4;
  const _Float16* A  = s_enc_h + (size_t)b * 2000 * 512 + (size_t)mt * 16 * 512;
  const _Float16* Bw = q_enc_h + (size_t)b * 112 * 512 + (size_t)wave * 16 * 512;
  v8f acc = splat8(0.f);
  for (int k0 = 0; k0 < 512; k0 += 32) {
    v16h a  = load_tileA16x32(A, 512, k0);
    v16h bb = load_tileB16x32(Bw, 512, k0);
    acc = wmma_f16(a, bb, acc);
  }
#pragma unroll
  for (int j = 0; j < 8; ++j) {
    int m = j + 8 * lh;
    alig[((size_t)(b * 2000 + mt * 16 + m)) * 112 + wave * 16 + nl] = acc[j];
  }
}

// masked softmax over q (len 100) per (b,k) row; writes f16 alig padded to 128 cols
__global__ void mask_softmax_kernel(const float* __restrict__ alig, const int* __restrict__ x,
                                    const int* __restrict__ s, _Float16* __restrict__ alig_h) {
  const int row = blockIdx.x;  // b*2000 + k
  const int b = row / 2000;
  const int tid = threadIdx.x;  // 128
  __shared__ float red[128];
  const int sm = (s[row] != 0);
  float a = 0.f;
  int msk = 0;
  if (tid < 100) {
    a = alig[(size_t)row * 112 + tid];
    msk = sm && (x[b * 100 + tid] != 0);
  }
  float val = (tid < 100 && msk) ? a : -1e10f;
  red[tid] = val;
  __syncthreads();
  for (int st = 64; st > 0; st >>= 1) {
    if (tid < st) red[tid] = fmaxf(red[tid], red[tid + st]);
    __syncthreads();
  }
  const float mx = red[0];
  __syncthreads();
  float e = (tid < 100) ? __expf(val - mx) : 0.f;
  red[tid] = e;
  __syncthreads();
  for (int st = 64; st > 0; st >>= 1) {
    if (tid < st) red[tid] += red[tid + st];
    __syncthreads();
  }
  const float sum = red[0];
  float o = (tid < 100) ? (e / sum) * (float)msk : 0.f;
  alig_h[(size_t)row * 128 + tid] = (_Float16)o;
}

// qT[b,d,q] = q_enc_h[b,q,d], q padded to 128
__global__ void transpose_q_kernel(const _Float16* __restrict__ q_enc_h,
                                   _Float16* __restrict__ qT) {
  size_t i = (size_t)blockIdx.x * 256 + threadIdx.x;
  if (i >= (size_t)16 * 512 * 128) return;
  int q = (int)(i & 127);
  int d = (int)((i >> 7) & 511);
  int b = (int)(i >> 16);
  qT[i] = (q < 112) ? q_enc_h[((size_t)b * 112 + q) * 512 + d] : (_Float16)0.f;
}

// soft_story[b,k,d] = sum_q alig[b,k,q] * q_enc[b,q,d]
__global__ __launch_bounds__(256) void soft_story_kernel(const _Float16* __restrict__ alig_h,
                                                         const _Float16* __restrict__ qT_h,
                                                         _Float16* __restrict__ soft_h) {
  const int b = blockIdx.y, mt = blockIdx.x;
  const int wave = threadIdx.x >> 5;
  const int lane = threadIdx.x & 31, nl = lane & 15, lh = lane >> 4;
  const _Float16* A = alig_h + (size_t)(b * 2000 + mt * 16) * 128;
#pragma unroll
  for (int i = 0; i < 4; ++i) {
    const int nt = wave * 4 + i;  // 32 N-tiles of 16 = 512 cols
    const _Float16* Bw = qT_h + (size_t)b * 512 * 128 + (size_t)nt * 16 * 128;
    v8f acc = splat8(0.f);
#pragma unroll
    for (int k0 = 0; k0 < 128; k0 += 32) {
      v16h a  = load_tileA16x32(A, 128, k0);
      v16h bb = load_tileB16x32(Bw, 128, k0);
      acc = wmma_f16(a, bb, acc);
    }
#pragma unroll
    for (int j = 0; j < 8; ++j) {
      int m = j + 8 * lh;
      soft_h[((size_t)(b * 2000 + mt * 16 + m)) * 512 + nt * 16 + nl] = (_Float16)acc[j];
    }
  }
}

// compare = cat([s*soft, (s-soft)^2], dim=-1) as f16 (LSTM input)
__global__ void build_compare_kernel(const _Float16* __restrict__ s_enc_h,
                                     const _Float16* __restrict__ soft_h,
                                     _Float16* __restrict__ cmp_in) {
  size_t i = (size_t)blockIdx.x * 256 + threadIdx.x;
  if (i >= (size_t)32000 * 512) return;
  size_t r = i >> 9;
  int d = (int)(i & 511);
  float a = (float)s_enc_h[i], b = (float)soft_h[i];
  cmp_in[r * 1024 + d] = (_Float16)(a * b);
  float dm = a - b;
  cmp_in[r * 1024 + 512 + d] = (_Float16)(dm * dm);
}

// ----------------------------------------------------------- tail kernels ---

__global__ void logits_kernel(const float* __restrict__ cw, const float* __restrict__ lw,
                              const float* __restrict__ lb, float* __restrict__ logits) {
  __shared__ float red[64];
  const int r = blockIdx.x, tid = threadIdx.x;  // 64 threads
  float p = 0.f;
  for (int d = tid; d < 512; d += 64) p += cw[(size_t)r * 512 + d] * lw[d];
  red[tid] = p;
  __syncthreads();
  for (int st = 32; st > 0; st >>= 1) {
    if (tid < st) red[tid] += red[tid + st];
    __syncthreads();
  }
  if (tid == 0) logits[r] = red[0] + lb[0];
}

__global__ void preds_kernel(const float* __restrict__ logits, float* __restrict__ out) {
  const int b = blockIdx.x, tid = threadIdx.x;  // 32 threads
  __shared__ float buf[20];
  __shared__ float mxs, sms;
  float v = 0.f;
  if (tid < 20) {
    v = logits[b * 20 + tid];
    buf[tid] = v;
  }
  __syncthreads();
  if (tid == 0) {
    float mx = -1e30f;
    for (int i = 0; i < 20; ++i) mx = fmaxf(mx, buf[i]);
    float s = 0.f;
    for (int i = 0; i < 20; ++i) s += __expf(buf[i] - mx);
    mxs = mx;
    sms = s;
  }
  __syncthreads();
  if (tid < 20) out[b * 20 + tid] = __expf(v - mxs) / sms;
}

// endl: hidden-size-1 LSTM over 320 steps of 512-dim input (scalar recurrence)
__global__ __launch_bounds__(256) void endl_kernel(const float* __restrict__ ew,
                                                   const float* __restrict__ Wih,  // (4,512)
                                                   const float* __restrict__ Whh,  // (4,1)
                                                   const float* __restrict__ bih,
                                                   const float* __restrict__ bhh,
                                                   float* __restrict__ out) {
  __shared__ float red[256];
  __shared__ float hc[2];
  const int tid = threadIdx.x;
  const int g = tid >> 6, j = tid & 63;  // 4 gates x 64 lanes
  if (tid == 0) { hc[0] = 0.f; hc[1] = 0.f; }
  __syncthreads();
  for (int t = 0; t < 320; ++t) {
    float p = 0.f;
    for (int d = j; d < 512; d += 64) p += ew[(size_t)t * 512 + d] * Wih[g * 512 + d];
    red[tid] = p;
    __syncthreads();
    for (int st = 32; st > 0; st >>= 1) {
      if (j < st) red[tid] += red[tid + st];
      __syncthreads();
    }
    if (tid == 0) {
      float h = hc[0], c = hc[1];
      float gi = red[0]   + h * Whh[0] + bih[0] + bhh[0];
      float gf = red[64]  + h * Whh[1] + bih[1] + bhh[1];
      float gg = red[128] + h * Whh[2] + bih[2] + bhh[2];
      float go = red[192] + h * Whh[3] + bih[3] + bhh[3];
      c = sigmf(gf) * c + sigmf(gi) * tanhf(gg);
      h = sigmf(go) * tanhf(c);
      hc[0] = h;
      hc[1] = c;
      out[t] = h;
    }
    __syncthreads();
  }
}

// ------------------------------------------------------------------ host ----

extern "C" void kernel_launch(void* const* d_in, const int* in_sizes, int n_in,
                              void* d_out, int out_size, void* d_ws, size_t ws_size,
                              hipStream_t stream) {
  (void)in_sizes; (void)n_in; (void)out_size; (void)ws_size;

  const int*   x       = (const int*)d_in[0];
  const int*   s       = (const int*)d_in[1];
  const float* emb     = (const float*)d_in[2];
  const float* ef_Wih  = (const float*)d_in[3];
  const float* ef_Whh  = (const float*)d_in[4];
  const float* ef_bih  = (const float*)d_in[5];
  const float* ef_bhh  = (const float*)d_in[6];
  const float* er_Wih  = (const float*)d_in[7];
  const float* er_Whh  = (const float*)d_in[8];
  const float* er_bih  = (const float*)d_in[9];
  const float* er_bhh  = (const float*)d_in[10];
  const float* c_Wih   = (const float*)d_in[11];
  const float* c_Whh   = (const float*)d_in[12];
  const float* c_bih   = (const float*)d_in[13];
  const float* c_bhh   = (const float*)d_in[14];
  const float* e_Wih   = (const float*)d_in[15];
  const float* e_Whh   = (const float*)d_in[16];
  const float* e_bih   = (const float*)d_in[17];
  const float* e_bhh   = (const float*)d_in[18];
  const float* l_Wih   = (const float*)d_in[19];
  const float* l_Whh   = (const float*)d_in[20];
  const float* l_bih   = (const float*)d_in[21];
  const float* l_bhh   = (const float*)d_in[22];
  const float* lin_w   = (const float*)d_in[23];
  const float* lin_b   = (const float*)d_in[24];
  float* out = (float*)d_out;

  char* ws = (char*)d_ws;
  size_t off = 0;
  auto alloc = [&](size_t bytes) -> void* {
    void* p = ws + off;
    off += (bytes + 255) & ~(size_t)255;
    return p;
  };

  _Float16* wihF = (_Float16*)alloc((size_t)1024 * 320 * 2);
  _Float16* whhF = (_Float16*)alloc((size_t)1024 * 256 * 2);
  _Float16* wihR = (_Float16*)alloc((size_t)1024 * 320 * 2);
  _Float16* whhR = (_Float16*)alloc((size_t)1024 * 256 * 2);
  _Float16* wihC = (_Float16*)alloc((size_t)2048 * 1024 * 2);
  _Float16* whhC = (_Float16*)alloc((size_t)2048 * 512 * 2);
  _Float16* wihE = (_Float16*)alloc((size_t)2048 * 1024 * 2);
  _Float16* whhE = (_Float16*)alloc((size_t)2048 * 512 * 2);
  float* bF = (float*)alloc(1024 * 4);
  float* bR = (float*)alloc(1024 * 4);
  float* bC = (float*)alloc(2048 * 4);
  float* bE = (float*)alloc(2048 * 4);
  _Float16* q_emb_h = (_Float16*)alloc((size_t)16 * 100 * 320 * 2);
  _Float16* s_emb_h = (_Float16*)alloc((size_t)320 * 100 * 320 * 2);
  _Float16* q_enc_h = (_Float16*)alloc((size_t)16 * 112 * 512 * 2);
  _Float16* qT_h    = (_Float16*)alloc((size_t)16 * 512 * 128 * 2);
  _Float16* s_enc_h = (_Float16*)alloc((size_t)320 * 100 * 512 * 2);
  float*    alig    = (float*)alloc((size_t)32000 * 112 * 4);
  _Float16* alig_h  = (_Float16*)alloc((size_t)32000 * 128 * 2);
  _Float16* soft_h  = (_Float16*)alloc((size_t)32000 * 512 * 2);
  _Float16* cmp_in  = (_Float16*)alloc((size_t)320 * 100 * 1024 * 2);
  float* cw     = (float*)alloc((size_t)320 * 512 * 4);
  float* ew     = (float*)alloc((size_t)320 * 512 * 4);
  float* logits = (float*)alloc(320 * 4);

  auto blocks = [](size_t n) { return (unsigned)((n + 255) / 256); };

  // 1) weights -> f16 (enc Wih padded 300->320), biases combined
  cvt_pad_kernel<<<blocks((size_t)1024 * 320), 256, 0, stream>>>(ef_Wih, wihF, 1024, 300, 320);
  cvt_pad_kernel<<<blocks((size_t)1024 * 256), 256, 0, stream>>>(ef_Whh, whhF, 1024, 256, 256);
  cvt_pad_kernel<<<blocks((size_t)1024 * 320), 256, 0, stream>>>(er_Wih, wihR, 1024, 300, 320);
  cvt_pad_kernel<<<blocks((size_t)1024 * 256), 256, 0, stream>>>(er_Whh, whhR, 1024, 256, 256);
  cvt_pad_kernel<<<blocks((size_t)2048 * 1024), 256, 0, stream>>>(c_Wih, wihC, 2048, 1024, 1024);
  cvt_pad_kernel<<<blocks((size_t)2048 * 512), 256, 0, stream>>>(c_Whh, whhC, 2048, 512, 512);
  cvt_pad_kernel<<<blocks((size_t)2048 * 1024), 256, 0, stream>>>(e_Wih, wihE, 2048, 1024, 1024);
  cvt_pad_kernel<<<blocks((size_t)2048 * 512), 256, 0, stream>>>(e_Whh, whhE, 2048, 512, 512);
  bias_add_kernel<<<blocks(1024), 256, 0, stream>>>(ef_bih, ef_bhh, bF, 1024);
  bias_add_kernel<<<blocks(1024), 256, 0, stream>>>(er_bih, er_bhh, bR, 1024);
  bias_add_kernel<<<blocks(2048), 256, 0, stream>>>(c_bih, c_bhh, bC, 2048);
  bias_add_kernel<<<blocks(2048), 256, 0, stream>>>(e_bih, e_bhh, bE, 2048);

  // 2) gather + normalize embeddings (E padded 300->320)
  gather_norm_kernel<<<1600, 128, 0, stream>>>(x, emb, q_emb_h);
  gather_norm_kernel<<<32000, 128, 0, stream>>>(s, emb, s_emb_h);

  // 3) BiLSTM encoders (query padded to 112 time rows, zeroed first)
  zero_f16_kernel<<<blocks((size_t)16 * 112 * 512), 256, 0, stream>>>(q_enc_h,
                                                                      (size_t)16 * 112 * 512);
  lstm_wmma_kernel<320, 256, 2, false><<<1, 256, 0, stream>>>(
      q_emb_h, wihF, whhF, bF, 100, 0, nullptr, q_enc_h, 512, 0, 112);
  lstm_wmma_kernel<320, 256, 2, false><<<1, 256, 0, stream>>>(
      q_emb_h, wihR, whhR, bR, 100, 1, nullptr, q_enc_h, 512, 256, 112);
  lstm_wmma_kernel<320, 256, 2, false><<<20, 256, 0, stream>>>(
      s_emb_h, wihF, whhF, bF, 100, 0, nullptr, s_enc_h, 512, 0, 100);
  lstm_wmma_kernel<320, 256, 2, false><<<20, 256, 0, stream>>>(
      s_emb_h, wihR, whhR, bR, 100, 1, nullptr, s_enc_h, 512, 256, 100);

  // 4) attention: scores, masked softmax, soft_story
  attn_score_kernel<<<dim3(125, 16), 256, 0, stream>>>(s_enc_h, q_enc_h, alig);
  mask_softmax_kernel<<<32000, 128, 0, stream>>>(alig, x, s, alig_h);
  transpose_q_kernel<<<blocks((size_t)16 * 512 * 128), 256, 0, stream>>>(q_enc_h, qT_h);
  soft_story_kernel<<<dim3(125, 16), 256, 0, stream>>>(alig_h, qT_h, soft_h);

  // 5) compare features, cmp/endr LSTMs with fused time-max-pool
  build_compare_kernel<<<blocks((size_t)32000 * 512), 256, 0, stream>>>(s_enc_h, soft_h, cmp_in);
  lstm_wmma_kernel<1024, 512, 4, true><<<20, 256, 0, stream>>>(
      cmp_in, wihC, whhC, bC, 100, 0, cw, nullptr, 512, 0, 1);
  lstm_wmma_kernel<1024, 512, 4, true><<<20, 256, 0, stream>>>(
      cmp_in, wihE, whhE, bE, 100, 0, ew, nullptr, 512, 0, 1);

  // 6) heads: preds (out[0:320]) and end_signal (out[320:640])
  logits_kernel<<<320, 64, 0, stream>>>(cw, lin_w, lin_b, logits);
  preds_kernel<<<16, 32, 0, stream>>>(logits, out);
  endl_kernel<<<1, 256, 0, stream>>>(ew, l_Wih, l_Whh, l_bih, l_bhh, out + 320);
}
